// GNet_77051713290258
// MI455X (gfx1250) — matrix operand order repaired and checked by reading, not compiled, not run on hardware
//
#include <hip/hip_runtime.h>
#include <hip/hip_bf16.h>

typedef __attribute__((ext_vector_type(16))) __bf16 v16bf;
typedef __attribute__((ext_vector_type(8)))  __bf16 v8bf;
typedef __attribute__((ext_vector_type(16))) float  v16f;
typedef __attribute__((ext_vector_type(8)))  float  v8f;

static constexpr int kN      = 50000;
static constexpr int kETrain = 800000;
static constexpr int kEPos   = 200000;
static constexpr int kENeg   = 200000;
static constexpr int kETest  = kEPos + kENeg;

static inline int cdiv(int a, int b) { return (a + b - 1) / b; }

// ---------------------------------------------------------------------------
// Degree / normalization (computed once; shared by all three layers)
// ---------------------------------------------------------------------------
__global__ void deg_init_kernel(float* __restrict__ deg, int n) {
    int i = blockIdx.x * blockDim.x + threadIdx.x;
    if (i < n) deg[i] = 1.0f;   // self-loop contributes 1
}

__global__ void deg_count_kernel(const int* __restrict__ dst, float* __restrict__ deg, int e) {
    int i = blockIdx.x * blockDim.x + threadIdx.x;
    if (i < e) atomicAdd(&deg[dst[i]], 1.0f);
}

__global__ void deg_rsqrt_kernel(const float* __restrict__ deg, float* __restrict__ dis, int n) {
    int i = blockIdx.x * blockDim.x + threadIdx.x;
    if (i < n) dis[i] = rsqrtf(deg[i]);
}

// ---------------------------------------------------------------------------
// W[K][N] f32  ->  Wt[N][K] bf16   (transposed so B fragments are K-contiguous)
// ---------------------------------------------------------------------------
__global__ void wt_bf16_kernel(const float* __restrict__ W, __bf16* __restrict__ Wt,
                               int K, int N) {
    int i = blockIdx.x * blockDim.x + threadIdx.x;
    if (i >= K * N) return;
    int k = i / N;
    int n = i - k * N;                 // coalesced read along N
    Wt[(size_t)n * K + k] = (__bf16)W[i];
}

// ---------------------------------------------------------------------------
// WMMA bf16 GEMM, LDS-free:  C[M,N] = A[M,K] @ W[K,N]  with W given as Wt[N][K] bf16.
// 256 threads = 8 waves. Block tile 128(M) x 64(N); wave w owns rows [w*16, w*16+16).
// Per K-step (32): each lane direct-loads its fragments (contiguous k order) from global:
//   A: row = wv*16 + (lane&15); K run  [hf*16,+16)                 -> 4x float4 + pk-cvt
//   B: col = nb*16 + (lane&15); K run  [hf*16,+16) of Wt           -> 2x 16B load
// 4x v_wmma_f32_16x16x32_bf16 per wave per K-step; fully unrolled over K.
// ---------------------------------------------------------------------------
template <int K, bool RELU_IN>
__global__ __launch_bounds__(256)
void gemm_wmma_kernel(const float* __restrict__ A, const __bf16* __restrict__ Wt,
                      float* __restrict__ C, int M, int N) {
    const int tid  = threadIdx.x;
    const int wv   = tid >> 5;       // wave id 0..7
    const int lane = tid & 31;
    const int hf   = lane >> 4;      // lane half
    const int r    = lane & 15;

    const int row_base = blockIdx.x * 128;
    const int col_base = blockIdx.y * 64;

    // A row for this lane's fragment (clamped so loads are unconditional)
    int arow = row_base + wv * 16 + r;
    if (arow >= M) arow = M - 1;
    const float* __restrict__ ap = A + (size_t)arow * K;

    // B fragment base pointers (Wt is [N][K] bf16; tiny, L0/L2 resident)
    const __bf16* __restrict__ bp0 = Wt + (size_t)(col_base + 0 * 16 + r) * K + hf * 16;
    const __bf16* __restrict__ bp1 = Wt + (size_t)(col_base + 1 * 16 + r) * K + hf * 16;
    const __bf16* __restrict__ bp2 = Wt + (size_t)(col_base + 2 * 16 + r) * K + hf * 16;
    const __bf16* __restrict__ bp3 = Wt + (size_t)(col_base + 3 * 16 + r) * K + hf * 16;

    v8f acc0 = {}, acc1 = {}, acc2 = {}, acc3 = {};

    #pragma unroll
    for (int k0 = 0; k0 < K; k0 += 32) {
        // ---- A fragment: one contiguous 16-float run, packed cvt to bf16 ----
        const float4 a0 = *(const float4*)(ap + k0 + hf * 16);
        const float4 a1 = *(const float4*)(ap + k0 + hf * 16 + 4);
        const float4 a2 = *(const float4*)(ap + k0 + hf * 16 + 8);
        const float4 a3 = *(const float4*)(ap + k0 + hf * 16 + 12);
        v16f av;
        av[0]  = a0.x; av[1]  = a0.y; av[2]  = a0.z; av[3]  = a0.w;
        av[4]  = a1.x; av[5]  = a1.y; av[6]  = a1.z; av[7]  = a1.w;
        av[8]  = a2.x; av[9]  = a2.y; av[10] = a2.z; av[11] = a2.w;
        av[12] = a3.x; av[13] = a3.y; av[14] = a3.z; av[15] = a3.w;
        if (RELU_IN) {
            #pragma unroll
            for (int j = 0; j < 16; ++j) av[j] = fmaxf(av[j], 0.0f);
        }
        const v16bf af = __builtin_convertvector(av, v16bf);

        // ---- B fragments: 16 contiguous bf16 each ----
        const v8bf b0lo = *(const v8bf*)(bp0 + k0);
        const v8bf b0hi = *(const v8bf*)(bp0 + k0 + 8);
        const v8bf b1lo = *(const v8bf*)(bp1 + k0);
        const v8bf b1hi = *(const v8bf*)(bp1 + k0 + 8);
        const v8bf b2lo = *(const v8bf*)(bp2 + k0);
        const v8bf b2hi = *(const v8bf*)(bp2 + k0 + 8);
        const v8bf b3lo = *(const v8bf*)(bp3 + k0);
        const v8bf b3hi = *(const v8bf*)(bp3 + k0 + 8);
        const v16bf bf0 = __builtin_shufflevector(b0lo, b0hi, 0,1,2,3,4,5,6,7,8,9,10,11,12,13,14,15);
        const v16bf bf1 = __builtin_shufflevector(b1lo, b1hi, 0,1,2,3,4,5,6,7,8,9,10,11,12,13,14,15);
        const v16bf bf2 = __builtin_shufflevector(b2lo, b2hi, 0,1,2,3,4,5,6,7,8,9,10,11,12,13,14,15);
        const v16bf bf3 = __builtin_shufflevector(b3lo, b3hi, 0,1,2,3,4,5,6,7,8,9,10,11,12,13,14,15);

        acc0 = __builtin_amdgcn_wmma_f32_16x16x32_bf16(false, af, false, bf0, (short)0, acc0, false, false);
        acc1 = __builtin_amdgcn_wmma_f32_16x16x32_bf16(false, af, false, bf1, (short)0, acc1, false, false);
        acc2 = __builtin_amdgcn_wmma_f32_16x16x32_bf16(false, af, false, bf2, (short)0, acc2, false, false);
        acc3 = __builtin_amdgcn_wmma_f32_16x16x32_bf16(false, af, false, bf3, (short)0, acc3, false, false);
    }

    // C layout: lane half hf, VGPR p -> row = wv*16 + hf*8 + p, col = 16*nb + r
    const int orow = row_base + wv * 16 + hf * 8;
    #pragma unroll
    for (int p = 0; p < 8; ++p) {
        const int gr = orow + p;
        if (gr < M) {
            const size_t base = (size_t)gr * N + col_base;
            C[base + 0 * 16 + r] = acc0[p];
            C[base + 1 * 16 + r] = acc1[p];
            C[base + 2 * 16 + r] = acc2[p];
            C[base + 3 * 16 + r] = acc3[p];
        }
    }
}

// ---------------------------------------------------------------------------
// Aggregation init: out[n][f] = h[n][f] * dis[n]^2 (self-loop) + bias[f]
// ---------------------------------------------------------------------------
__global__ void agg_init_kernel(const float* __restrict__ h, const float* __restrict__ dis,
                                const float* __restrict__ bias, float* __restrict__ out,
                                int total, int l2F) {
    int i = blockIdx.x * blockDim.x + threadIdx.x;
    if (i >= total) return;
    int node = i >> l2F;
    int f    = i & ((1 << l2F) - 1);
    float d  = dis[node];
    out[i] = h[i] * d * d + bias[f];
}

// ---------------------------------------------------------------------------
// Edge scatter-add: out[dst] += h[src] * dis[src]*dis[dst]
// One thread per (edge, 4 contiguous floats): float4 gather + 4 f32 atomics.
// ---------------------------------------------------------------------------
__global__ void agg_edges_kernel(const float* __restrict__ h, const int* __restrict__ src,
                                 const int* __restrict__ dst, const float* __restrict__ dis,
                                 float* __restrict__ out, int total, int l2per, int F) {
    int i = blockIdx.x * blockDim.x + threadIdx.x;
    if (i >= total) return;
    int e = i >> l2per;
    int c = i & ((1 << l2per) - 1);
    int f = c << 2;
    int s = src[e];
    int d = dst[e];
    float norm = dis[s] * dis[d];
    const float4 hv = *(const float4*)(h + (size_t)s * F + f);
    float* o = out + (size_t)d * F + f;
    atomicAdd(o + 0, hv.x * norm);
    atomicAdd(o + 1, hv.y * norm);
    atomicAdd(o + 2, hv.z * norm);
    atomicAdd(o + 3, hv.w * norm);
}

// ---------------------------------------------------------------------------
// Edge scoring: one wave32 per test edge, 64-float dot product.
// ---------------------------------------------------------------------------
__global__ __launch_bounds__(256)
void edge_score_kernel(const float* __restrict__ h, const int* __restrict__ pos,
                       const int* __restrict__ neg, float* __restrict__ out) {
    int e    = blockIdx.x * 8 + (threadIdx.x >> 5);
    int lane = threadIdx.x & 31;
    if (e >= kETest) return;
    int s, d;
    if (e < kEPos) { s = pos[e];          d = pos[kEPos + e]; }
    else           { int e2 = e - kEPos;  s = neg[e2]; d = neg[kENeg + e2]; }
    const float2 a = *(const float2*)(h + (size_t)s * 64 + lane * 2);
    const float2 b = *(const float2*)(h + (size_t)d * 64 + lane * 2);
    float p = a.x * b.x + a.y * b.y;
    #pragma unroll
    for (int off = 16; off > 0; off >>= 1) p += __shfl_xor(p, off, 32);
    if (lane == 0) out[e] = p;
}

// ---------------------------------------------------------------------------
extern "C" void kernel_launch(void* const* d_in, const int* in_sizes, int n_in,
                              void* d_out, int out_size, void* d_ws, size_t ws_size,
                              hipStream_t stream) {
    const float* x   = (const float*)d_in[0];
    const int*   tr  = (const int*)d_in[1];
    const int*   pos = (const int*)d_in[2];
    const int*   neg = (const int*)d_in[3];
    const float* W1  = (const float*)d_in[4];
    const float* b1  = (const float*)d_in[5];
    const float* W2  = (const float*)d_in[6];
    const float* b2  = (const float*)d_in[7];
    const float* W3  = (const float*)d_in[8];
    const float* b3  = (const float*)d_in[9];

    const int* src = tr;             // train_pos_edge_index[0]
    const int* dst = tr + kETrain;   // train_pos_edge_index[1]

    // Workspace carve-out (256B aligned)
    char* ws = (char*)d_ws;
    size_t cur = 0;
    auto carve = [&](size_t bytes) -> void* {
        cur = (cur + 255) & ~(size_t)255;
        void* p = ws + cur;
        cur += bytes;
        return p;
    };
    float*  deg   = (float*)carve((size_t)kN * 4);
    float*  dis   = (float*)carve((size_t)kN * 4);
    __bf16* W1t   = (__bf16*)carve((size_t)256 * 128 * 2);   // [N=256][K=128]
    __bf16* W2t   = (__bf16*)carve((size_t)128 * 256 * 2);   // [N=128][K=256]
    __bf16* W3t   = (__bf16*)carve((size_t)64 * 128 * 2);    // [N=64][K=128]
    float*  hbuf  = (float*)carve((size_t)kN * 256 * 4);     // GEMM output scratch (max F)
    float*  abuf1 = (float*)carve((size_t)kN * 256 * 4);     // layer 1 aggregated
    float*  abuf2 = (float*)carve((size_t)kN * 128 * 4);     // layer 2 aggregated
    float*  abuf3 = (float*)carve((size_t)kN * 64 * 4);      // layer 3 aggregated (final h)
    (void)ws_size; (void)in_sizes; (void)n_in; (void)out_size;

    const int T = 256;

    // Degrees + rsqrt norm (shared across layers)
    deg_init_kernel<<<cdiv(kN, T), T, 0, stream>>>(deg, kN);
    deg_count_kernel<<<cdiv(kETrain, T), T, 0, stream>>>(dst, deg, kETrain);
    deg_rsqrt_kernel<<<cdiv(kN, T), T, 0, stream>>>(deg, dis, kN);

    // Weight conversion to transposed bf16
    wt_bf16_kernel<<<cdiv(128 * 256, T), T, 0, stream>>>(W1, W1t, 128, 256);
    wt_bf16_kernel<<<cdiv(256 * 128, T), T, 0, stream>>>(W2, W2t, 256, 128);
    wt_bf16_kernel<<<cdiv(128 * 64, T), T, 0, stream>>>(W3, W3t, 128, 64);

    // ---- Layer 1: 128 -> 256 ----
    gemm_wmma_kernel<128, false><<<dim3(cdiv(kN, 128), 256 / 64), 256, 0, stream>>>(x, W1t, hbuf, kN, 256);
    agg_init_kernel<<<cdiv(kN * 256, T), T, 0, stream>>>(hbuf, dis, b1, abuf1, kN * 256, 8);
    agg_edges_kernel<<<cdiv(kETrain * (256 / 4), T), T, 0, stream>>>(hbuf, src, dst, dis, abuf1,
                                                                     kETrain * (256 / 4), 6, 256);

    // ---- Layer 2: 256 -> 128 (ReLU fused into A load) ----
    gemm_wmma_kernel<256, true><<<dim3(cdiv(kN, 128), 128 / 64), 256, 0, stream>>>(abuf1, W2t, hbuf, kN, 128);
    agg_init_kernel<<<cdiv(kN * 128, T), T, 0, stream>>>(hbuf, dis, b2, abuf2, kN * 128, 7);
    agg_edges_kernel<<<cdiv(kETrain * (128 / 4), T), T, 0, stream>>>(hbuf, src, dst, dis, abuf2,
                                                                     kETrain * (128 / 4), 5, 128);

    // ---- Layer 3: 128 -> 64 (ReLU fused into A load; no ReLU on output) ----
    gemm_wmma_kernel<128, true><<<dim3(cdiv(kN, 128), 64 / 64), 256, 0, stream>>>(abuf2, W3t, hbuf, kN, 64);
    agg_init_kernel<<<cdiv(kN * 64, T), T, 0, stream>>>(hbuf, dis, b3, abuf3, kN * 64, 6);
    agg_edges_kernel<<<cdiv(kETrain * (64 / 4), T), T, 0, stream>>>(hbuf, src, dst, dis, abuf3,
                                                                    kETrain * (64 / 4), 4, 64);

    // ---- Edge scoring: 400k dot products over 64 features ----
    edge_score_kernel<<<cdiv(kETest, 8), 256, 0, stream>>>(abuf3, pos, neg, (float*)d_out);
}
